// DiverseSampler_10634339025303
// MI455X (gfx1250) — matrix operand order repaired and checked by reading, not compiled, not run on hardware
//
#include <hip/hip_runtime.h>
#include <hip/hip_bf16.h>
#include <stdint.h>

// Problem constants (match the reference)
#define BB 8
#define NN 4096
#define DD 1024
#define KK 256

typedef __attribute__((ext_vector_type(16))) __bf16 v16bf;
typedef __attribute__((ext_vector_type(8)))  __bf16 v8bf;
typedef __attribute__((ext_vector_type(8)))  float  v8f;

__device__ __forceinline__ uint64_t splitmix64(uint64_t z) {
    z += 0x9E3779B97F4A7C15ull;
    z = (z ^ (z >> 30)) * 0xBF58476D1CE4E5B9ull;
    z = (z ^ (z >> 27)) * 0x94D049BB133111EBull;
    return z ^ (z >> 31);
}

// ---------------------------------------------------------------------------
// Kernel 1: convert x (f32) -> bf16 workspace copy, and per-token squared norms
// grid = B*N blocks, 256 threads
// ---------------------------------------------------------------------------
__global__ void kpp_prep(const float* __restrict__ x,
                         __bf16* __restrict__ xb,
                         float* __restrict__ xn) {
    const int row = blockIdx.x;             // 0 .. B*N-1
    const int t   = threadIdx.x;            // 0 .. 255
    const float* src = x + (size_t)row * DD;
    __bf16*      dst = xb + (size_t)row * DD;

    float s = 0.0f;
    #pragma unroll
    for (int i = t; i < DD; i += 256) {
        float v = src[i];
        dst[i]  = (__bf16)v;
        s += v * v;
    }
    __shared__ float red[256];
    red[t] = s;
    __syncthreads();
    for (int off = 128; off > 0; off >>= 1) {
        if (t < off) red[t] += red[t + off];
        __syncthreads();
    }
    if (t == 0) xn[row] = red[0];
}

// ---------------------------------------------------------------------------
// Kernel 2: persistent per-batch-row k-means++ chain.
// grid = B blocks, 1024 threads (32 wave32s).
// Distance pass uses v_wmma_f32_16x16x32_bf16: A = 16 token rows x 32-chunk of
// D, B = the current center chunk broadcast into all 16 columns; column 0 of C
// (lanes 0/16, VGPRs 0-7) carries the 16 dot products. Accumulate over D with
// 32 chained WMMAs per 16-row tile.
// ---------------------------------------------------------------------------
__global__ void
__launch_bounds__(1024)
kpp_chain(const __bf16* __restrict__ xb,
          const float* __restrict__ xn,
          int* __restrict__ idx) {
    const int b    = blockIdx.x;
    const int t    = threadIdx.x;       // 0..1023
    const int lane = t & 31;
    const int wave = t >> 5;            // 0..31

    __shared__ __align__(32) __bf16 s_c[DD];   // current center (bf16)  2 KB
    __shared__ float s_d[NN];                  // running min distance  16 KB
    __shared__ float s_scan[1024];             // block scan             4 KB
    __shared__ int   s_cur, s_sel;
    __shared__ float s_cnorm, s_u;

    // init distances to +inf
    for (int i = t; i < NN; i += 1024) s_d[i] = 3.402823e38f;

    const uint64_t seedb = 0x9E3779B97F4A7C15ull * (uint64_t)(b + 1) ^ 42ull;
    if (t == 0) {
        int first = (int)(splitmix64(seedb) % (uint64_t)NN);
        s_cur = first;
        idx[b * KK + 0] = first;
    }
    __syncthreads();

    // A-fragment: lanes 0-15 hold K 0-15, lanes 16-31 hold K 16-31
    // (contiguous). B-fragment: lanes 0-15 hold
    // K 0-15, lanes 16-31 hold K 16-31 (contiguous).
    const int aoff  = (lane & 16) ? 16 : 0;
    const int boff  = (lane & 16) ? 16 : 0;
    const int arowm = lane & 15;        // M row within tile held by this lane
    const int mhalf = (lane & 16) ? 8 : 0;

    for (int k = 1; k < KK; ++k) {
        const int cur = s_cur;
        // stage current center into LDS + its norm
        for (int i = t; i < DD; i += 1024)
            s_c[i] = xb[((size_t)(b * NN + cur)) * DD + i];
        if (t == 0) s_cnorm = xn[b * NN + cur];
        __syncthreads();
        const float cn = s_cnorm;

        // ---- WMMA distance update: each wave owns 8 of the 256 row-tiles ----
        for (int tile = wave; tile < NN / 16; tile += 32) {
            v8f acc = {};
            const __bf16* arow =
                xb + ((size_t)(b * NN) + tile * 16 + arowm) * DD;
            #pragma unroll 4
            for (int kc = 0; kc < DD; kc += 32) {
                union { v16bf v; v8bf h[2]; } ua;
                ua.h[0] = *(const v8bf*)(arow + kc + aoff);
                ua.h[1] = *(const v8bf*)(arow + kc + aoff + 8);
                v16bf bv = *(const v16bf*)(s_c + kc + boff);
                acc = __builtin_amdgcn_wmma_f32_16x16x32_bf16(
                    false, ua.v, false, bv, (short)0, acc, false, false);
            }
            // column 0 of C: lane 0 -> rows m=0..7, lane 16 -> rows m=8..15
            if ((lane & 15) == 0) {
                const int mbase = tile * 16 + mhalf;
                #pragma unroll
                for (int r = 0; r < 8; ++r) {
                    const int row = mbase + r;
                    float nd2 = xn[b * NN + row] - 2.0f * acc[r] + cn;
                    float nd  = __builtin_sqrtf(fmaxf(nd2, 0.0f));
                    s_d[row]  = fminf(s_d[row], nd);
                }
            }
        }
        __syncthreads();

        // ---- categorical sample proportional to d (torch.multinomial) ----
        const float d0 = s_d[t * 4 + 0];
        const float d1 = s_d[t * 4 + 1];
        const float d2 = s_d[t * 4 + 2];
        const float d3 = s_d[t * 4 + 3];
        const float part = d0 + d1 + d2 + d3;
        s_scan[t] = part;
        __syncthreads();
        // Hillis-Steele inclusive scan over 1024 partials
        for (int off = 1; off < 1024; off <<= 1) {
            float v = (t >= off) ? s_scan[t - off] : 0.0f;
            __syncthreads();
            s_scan[t] += v;
            __syncthreads();
        }
        const float total = s_scan[1023];
        if (t == 0) {
            uint64_t h = splitmix64(seedb + 0x632BE59BD9B4E019ull * (uint64_t)k);
            s_u  = (float)((h >> 40) & 0xFFFFFF) * (1.0f / 16777216.0f) * total;
            s_sel = 0x7FFFFFFF;
        }
        __syncthreads();
        const float u    = s_u;
        const float incl = s_scan[t];
        const float excl = incl - part;
        if (part > 0.0f && u >= excl && u < incl) {
            float run = excl;
            run += d0; if (u < run) { atomicMin(&s_sel, t * 4 + 0); goto done; }
            run += d1; if (u < run) { atomicMin(&s_sel, t * 4 + 1); goto done; }
            run += d2; if (u < run) { atomicMin(&s_sel, t * 4 + 2); goto done; }
            run += d3; if (u < run) { atomicMin(&s_sel, t * 4 + 3); }
        }
        done:
        __syncthreads();
        if (t == 0) {
            int sel = s_sel;
            if (sel >= NN) sel = NN - 1;   // u landed on total (fp edge)
            s_cur = sel;
            idx[b * KK + k] = sel;
        }
        __syncthreads();
    }
}

// ---------------------------------------------------------------------------
// Kernel 3: gather chosen tokens (f32) into d_out + trailing scalar 0.0
// grid = B*K blocks, 256 threads, one float4 per thread
// ---------------------------------------------------------------------------
__global__ void kpp_gather(const float* __restrict__ x,
                           const int* __restrict__ idx,
                           float* __restrict__ out) {
    const int blk = blockIdx.x;       // 0..B*K-1
    const int b   = blk >> 8;         // K = 256
    const int t   = threadIdx.x;      // 0..255 -> one float4 each (D=1024)
    const int row = idx[blk];
    const float4* src = (const float4*)(x + ((size_t)b * NN + row) * DD);
    float4*       dst = (float4*)(out + (size_t)blk * DD);
    dst[t] = src[t];
    if (blk == 0 && t == 0)
        out[(size_t)BB * KK * DD] = 0.0f;  // second tuple element
}

extern "C" void kernel_launch(void* const* d_in, const int* in_sizes, int n_in,
                              void* d_out, int out_size, void* d_ws, size_t ws_size,
                              hipStream_t stream) {
    (void)in_sizes; (void)n_in; (void)out_size; (void)ws_size;
    const float* x = (const float*)d_in[0];
    float* out = (float*)d_out;

    // workspace layout: bf16 copy of x | per-token squared norms | indices
    char* ws = (char*)d_ws;
    const size_t xb_bytes = (size_t)BB * NN * DD * sizeof(__bf16);   // 64 MB
    const size_t xn_bytes = (size_t)BB * NN * sizeof(float);         // 128 KB
    __bf16* xb  = (__bf16*)ws;
    float*  xn  = (float*)(ws + xb_bytes);
    int*    idx = (int*)(ws + xb_bytes + xn_bytes);

    kpp_prep  <<<BB * NN, 256, 0, stream>>>(x, xb, xn);
    kpp_chain <<<BB,      1024, 0, stream>>>(xb, xn, idx);
    kpp_gather<<<BB * KK, 256, 0, stream>>>(x, idx, out);
}
